// BRBboxHead_37280316129469
// MI455X (gfx1250) — matrix operand order repaired and checked by reading, not compiled, not run on hardware
//
#include <hip/hip_runtime.h>
#include <hip/hip_bf16.h>

typedef __attribute__((ext_vector_type(16))) __bf16 v16bf;
typedef __attribute__((ext_vector_type(8)))  float  v8f;

#define BN_EPS 1e-5f

__device__ __forceinline__ unsigned short f2bf(float x) {
    __bf16 h = (__bf16)x;
    return __builtin_bit_cast(unsigned short, h);
}

// A-fragment for v_wmma_*_16x16x32_bf16 from a global fp32 weight row,
// scaled (BN fold) and converted to bf16.
// Layout (contiguous K, matches B): lane half h: elems 0..15 = K kbase+16h..+15.
__device__ __forceinline__ v16bf load_a_f32(const float* __restrict__ row,
                                            int kbase, int h, float scale) {
    const float* p0 = row + kbase + h * 16;
    const float* p1 = row + kbase + h * 16 + 8;
    float4 x0 = *(const float4*)(p0);
    float4 x1 = *(const float4*)(p0 + 4);
    float4 y0 = *(const float4*)(p1);
    float4 y1 = *(const float4*)(p1 + 4);
    v16bf a;
    a[0]  = (__bf16)(x0.x * scale); a[1]  = (__bf16)(x0.y * scale);
    a[2]  = (__bf16)(x0.z * scale); a[3]  = (__bf16)(x0.w * scale);
    a[4]  = (__bf16)(x1.x * scale); a[5]  = (__bf16)(x1.y * scale);
    a[6]  = (__bf16)(x1.z * scale); a[7]  = (__bf16)(x1.w * scale);
    a[8]  = (__bf16)(y0.x * scale); a[9]  = (__bf16)(y0.y * scale);
    a[10] = (__bf16)(y0.z * scale); a[11] = (__bf16)(y0.w * scale);
    a[12] = (__bf16)(y1.x * scale); a[13] = (__bf16)(y1.y * scale);
    a[14] = (__bf16)(y1.z * scale); a[15] = (__bf16)(y1.w * scale);
    return a;
}

// B-fragment: lane (n = lane%16, h = lane/16) reads 16 contiguous bf16 at
// K = s*32 + 16h from the point-major LDS tile (two ds_load_b128).
__device__ __forceinline__ v16bf load_b_lds(const unsigned short* p) {
    union { uint4 q[2]; v16bf v; } u;
    u.q[0] = *(const uint4*)(p);
    u.q[1] = *(const uint4*)(p + 8);
    return u.v;
}

#define WMMA_BF16(A, B, C) \
    __builtin_amdgcn_wmma_f32_16x16x32_bf16(false, (A), false, (B), (short)0, (C), false, false)

__global__ __launch_bounds__(256, 1)
void brb_head_fused(const float* __restrict__ X,      // [32][256][4096]
                    const float* __restrict__ cang,   // [32][4096]
                    const float* __restrict__ cdist,  // [32][4096][6]
                    const float* __restrict__ W0, const float* __restrict__ b0,
                    const float* __restrict__ g0, const float* __restrict__ be0,
                    const float* __restrict__ m0, const float* __restrict__ v0,
                    const float* __restrict__ W1, const float* __restrict__ b1,
                    const float* __restrict__ g1, const float* __restrict__ be1,
                    const float* __restrict__ m1, const float* __restrict__ v1,
                    const float* __restrict__ Wc, const float* __restrict__ bc,
                    const float* __restrict__ Wr, const float* __restrict__ br,
                    float* __restrict__ out) {
    constexpr int CIN = 256, N = 4096, C1 = 128, TN = 128;
    constexpr int SXROW = CIN + 8;   // 264 bf16; row stride 132 dwords -> bank step 4
    constexpr int SHROW = C1 + 8;    // 136 bf16; row stride 68 dwords  -> bank step 4
    constexpr int SX_BYTES = TN * SXROW * 2;   // 67584
    constexpr int SH_BYTES = TN * SHROW * 2;   // 34816

    // Aliased LDS arena: [sX | sH0], later [sH1, sOut | sH0]
    __shared__ __align__(16) unsigned char smem[SX_BYTES + SH_BYTES];
    unsigned short* sX   = (unsigned short*)smem;               // [TN][264], phase 0-1
    unsigned short* sH0  = (unsigned short*)(smem + SX_BYTES);  // [TN][136]
    unsigned short* sH1  = (unsigned short*)smem;               // [TN][136], aliases dead sX
    float*          sOut = (float*)(smem + 36864);              // [TN][32] f32 head staging
    __shared__ float sB0[128], sB1[128], sBh[32];

    const int tid  = threadIdx.x;
    const int lane = tid & 31;
    const int w    = tid >> 5;      // 8 waves
    const int l16  = lane & 15;
    const int h    = lane >> 4;
    const int b    = blockIdx.x >> 5;          // 32 N-tiles per batch
    const int n0   = (blockIdx.x & 31) * TN;

    // ---- phase 0: fused biases + stage X tile as bf16 (point-major) ----
    if (tid < 128) {
        float s0 = g0[tid] * rsqrtf(v0[tid] + BN_EPS);
        sB0[tid] = (b0[tid] - m0[tid]) * s0 + be0[tid];
        float s1 = g1[tid] * rsqrtf(v1[tid] + BN_EPS);
        sB1[tid] = (b1[tid] - m1[tid]) * s1 + be1[tid];
    }
    if (tid < 32) sBh[tid] = (tid < 18) ? bc[tid] : ((tid < 25) ? br[tid - 18] : 0.0f);

    const float* Xb = X + ((size_t)b * CIN) * N + n0;
    #pragma unroll 4
    for (int it = 0; it < 32; ++it) {           // 256 rows x 32 float4 cols
        int idx = it * 256 + tid;
        int c   = idx >> 5;
        int nn  = (idx & 31) * 4;
        float4 v = *(const float4*)(Xb + (size_t)c * N + nn);
        sX[(nn + 0) * SXROW + c] = f2bf(v.x);
        sX[(nn + 1) * SXROW + c] = f2bf(v.y);
        sX[(nn + 2) * SXROW + c] = f2bf(v.z);
        sX[(nn + 3) * SXROW + c] = f2bf(v.w);
    }
    __syncthreads();

    // ---- layer 0: H0 = relu(scale0*W0 @ X + b0')  (K=256, wave w -> rows 16w..16w+15)
    {
        int mrow = w * 16 + l16;
        float sc = g0[mrow] * rsqrtf(v0[mrow] + BN_EPS);
        const float* Wrow = W0 + mrow * CIN;
        v16bf a[8];
        #pragma unroll
        for (int s = 0; s < 8; ++s) a[s] = load_a_f32(Wrow, s * 32, h, sc);
        for (int j = 0; j < 8; ++j) {
            int n = j * 16 + l16;
            const unsigned short* bp = sX + n * SXROW + h * 16;
            v8f acc = {};
            #pragma unroll
            for (int s = 0; s < 8; ++s) {
                v16bf bb = load_b_lds(bp + s * 32);
                acc = WMMA_BF16(a[s], bb, acc);
            }
            uint4 pk; __bf16* ph = (__bf16*)&pk;  // D: VGPR r -> M = 16w + 8h + r (contig)
            #pragma unroll
            for (int r = 0; r < 8; ++r)
                ph[r] = (__bf16)fmaxf(acc[r] + sB0[w * 16 + 8 * h + r], 0.0f);
            *(uint4*)(sH0 + n * SHROW + w * 16 + 8 * h) = pk;
        }
    }
    __syncthreads();

    // ---- layer 1: H1 = relu(scale1*W1 @ H0 + b1')  (K=128; sH1 reuses dead sX arena)
    {
        int mrow = w * 16 + l16;
        float sc = g1[mrow] * rsqrtf(v1[mrow] + BN_EPS);
        const float* Wrow = W1 + mrow * C1;
        v16bf a[4];
        #pragma unroll
        for (int s = 0; s < 4; ++s) a[s] = load_a_f32(Wrow, s * 32, h, sc);
        for (int j = 0; j < 8; ++j) {
            int n = j * 16 + l16;
            const unsigned short* bp = sH0 + n * SHROW + h * 16;
            v8f acc = {};
            #pragma unroll
            for (int s = 0; s < 4; ++s) {
                v16bf bb = load_b_lds(bp + s * 32);
                acc = WMMA_BF16(a[s], bb, acc);
            }
            uint4 pk; __bf16* ph = (__bf16*)&pk;
            #pragma unroll
            for (int r = 0; r < 8; ++r)
                ph[r] = (__bf16)fmaxf(acc[r] + sB1[w * 16 + 8 * h + r], 0.0f);
            *(uint4*)(sH1 + n * SHROW + w * 16 + 8 * h) = pk;
        }
    }
    __syncthreads();

    // ---- heads: [Wc;Wr] @ H1 -> 25 (pad 32) output rows; 2 m-tiles x 8 n-subtiles
    {
        int mh   = w & 1;
        int jb   = w >> 1;
        int mrow = mh * 16 + l16;                          // 0..31
        const float* Wrow = (mrow < 18) ? (Wc + mrow * C1)
                          : ((mrow < 25) ? (Wr + (mrow - 18) * C1) : Wc);
        float sc = (mrow < 25) ? 1.0f : 0.0f;              // zero the pad rows
        v16bf a[4];
        #pragma unroll
        for (int s = 0; s < 4; ++s) a[s] = load_a_f32(Wrow, s * 32, h, sc);
        #pragma unroll
        for (int jj = 0; jj < 2; ++jj) {
            int n = (jb + jj * 4) * 16 + l16;
            const unsigned short* bp = sH1 + n * SHROW + h * 16;
            v8f acc = {};
            #pragma unroll
            for (int s = 0; s < 4; ++s) {
                v16bf bb = load_b_lds(bp + s * 32);
                acc = WMMA_BF16(a[s], bb, acc);
            }
            float* op = sOut + n * 32 + mh * 16 + 8 * h;
            *(float4*)(op)     = make_float4(acc[0], acc[1], acc[2], acc[3]);
            *(float4*)(op + 4) = make_float4(acc[4], acc[5], acc[6], acc[7]);
        }
    }
    __syncthreads();

    // ---- epilogue: transpose to [B,N,C], add head bias + coarse residuals ----
    const size_t gbase = (size_t)b * N + n0;
    float* angO = out + (size_t)32 * 4096 * 18;
    float* disO = angO + (size_t)32 * 4096;
    for (int idx = tid; idx < TN * 25; idx += 256) {
        int n = idx / 25;
        int m = idx - n * 25;
        float v = sOut[n * 32 + m] + sBh[m];
        size_t gn = gbase + n;
        if (m < 18)            out[gn * 18 + m] = v;                       // sem_scores
        else if (m == 18)      angO[gn] = cang[gn] + v;                    // refined_angle
        else { int d = m - 19; disO[gn * 6 + d] = cdist[gn * 6 + d] + v; } // refined_distance
    }
}

extern "C" void kernel_launch(void* const* d_in, const int* in_sizes, int n_in,
                              void* d_out, int out_size, void* d_ws, size_t ws_size,
                              hipStream_t stream) {
    (void)in_sizes; (void)n_in; (void)d_ws; (void)ws_size; (void)out_size;
    const float* X     = (const float*)d_in[0];
    const float* cang  = (const float*)d_in[1];
    const float* cdist = (const float*)d_in[2];
    const float* W0 = (const float*)d_in[3];
    const float* b0 = (const float*)d_in[4];
    const float* g0 = (const float*)d_in[5];
    const float* be0= (const float*)d_in[6];
    const float* m0 = (const float*)d_in[7];
    const float* v0 = (const float*)d_in[8];
    const float* W1 = (const float*)d_in[9];
    const float* b1 = (const float*)d_in[10];
    const float* g1 = (const float*)d_in[11];
    const float* be1= (const float*)d_in[12];
    const float* m1 = (const float*)d_in[13];
    const float* v1 = (const float*)d_in[14];
    const float* Wc = (const float*)d_in[15];
    const float* bc = (const float*)d_in[16];
    const float* Wr = (const float*)d_in[17];
    const float* br = (const float*)d_in[18];

    dim3 grid(32 * 32);   // (batch, 128-point tile)
    dim3 block(256);      // 8 wave32
    brb_head_fused<<<grid, block, 0, stream>>>(X, cang, cdist,
                                               W0, b0, g0, be0, m0, v0,
                                               W1, b1, g1, be1, m1, v1,
                                               Wc, bc, Wr, br, (float*)d_out);
}
